// LSTM_42494406426768
// MI455X (gfx1250) — compile-verified, matrix operand order checked by reading, not run on hardware
//
#include <hip/hip_runtime.h>
#include <hip/hip_bf16.h>
#include <cstdint>
#include <cstddef>

typedef __bf16 bf16;
typedef __attribute__((ext_vector_type(8)))  __bf16 v8bf;
typedef __attribute__((ext_vector_type(16))) __bf16 v16bf;
typedef __attribute__((ext_vector_type(8)))  float  v8f;

#define B_   512
#define T_   40
#define E_   300
#define EP_  320        // E padded to multiple of 32 for K of wmma
#define H_   512
#define G3_  1536       // 3*H
#define BT_  (B_ * T_)  // 20480

// ---------------------------------------------------------------------------
// Fragment helpers per CDNA5 ISA 7.12.2 (wave32):
//  A 16x32 bf16: lane<16 -> row M=lane, elems 0..7 = K k0..k0+7, 8..15 = k0+16..k0+23
//                lane>=16 -> same rows, K shifted by +8.  => two 16B loads.
//  B 32x16 bf16: symmetric with N = lane&15 reading row-major [N][K].
//  C/D 16x16 f32: elem r at (M = r + 8*(lane>=16), N = lane&15).
// ---------------------------------------------------------------------------
__device__ __forceinline__ v16bf load_frag(const bf16* p) {
  v8bf lo = *(const v8bf*)(p);
  v8bf hi = *(const v8bf*)(p + 16);
  v16bf r;
#pragma unroll
  for (int i = 0; i < 8; ++i) { r[i] = lo[i]; r[i + 8] = hi[i]; }
  return r;
}

__device__ __forceinline__ float sigmoidf_(float x) {
  return 1.0f / (1.0f + __expf(-x));
}

// ---------------------------------------------------------------------------
// Cast f32 -> bf16 with optional column zero-padding (weights).
// ---------------------------------------------------------------------------
__global__ void castpad_kernel(const float* __restrict__ src, bf16* __restrict__ dst,
                               int rows, int scols, int dcols) {
  int gid = blockIdx.x * blockDim.x + threadIdx.x;
  int total = rows * dcols;
  if (gid >= total) return;
  int r = gid / dcols, c = gid % dcols;
  float v = (c < scols) ? src[(size_t)r * scols + c] : 0.0f;
  dst[gid] = (bf16)v;
}

// ---------------------------------------------------------------------------
// Embedding gather + cast to bf16, K-padded 300 -> 320.  X: [BT][EP]
// ---------------------------------------------------------------------------
__global__ void gather_kernel(const int* __restrict__ inds, const float* __restrict__ emb,
                              bf16* __restrict__ X) {
  int gid = blockIdx.x * blockDim.x + threadIdx.x;
  if (gid >= BT_ * EP_) return;
  int r = gid / EP_, e = gid % EP_;
  int ind = inds[r];
  float v = (e < E_) ? emb[(size_t)ind * E_ + e] : 0.0f;
  X[gid] = (bf16)v;
}

// ---------------------------------------------------------------------------
// Input-gate GEMM: GI[BT][G3] = X[BT][EP] @ W[G3][EP]^T + b_ih  (stored bf16)
// 256 threads = 8 waves; wave w -> rows m0+w*16..+15, block covers 128x64 tile.
// ---------------------------------------------------------------------------
__global__ __launch_bounds__(256) void gemm_gi_kernel(
    const bf16* __restrict__ X, const bf16* __restrict__ W,
    const float* __restrict__ bih, bf16* __restrict__ GI) {
  const int lane = threadIdx.x & 31;
  const int wave = threadIdx.x >> 5;
  const int ln   = lane & 15;
  const int kx   = (lane >> 4) << 3;
  const int m0   = blockIdx.y * 128 + wave * 16;
  const int n0   = blockIdx.x * 64;

  v8f acc[4] = {};
  const bf16* xrow = X + (size_t)(m0 + ln) * EP_;

  for (int k0 = 0; k0 < EP_; k0 += 32) {
    const int k = k0 + kx;
    v16bf a = load_frag(xrow + k);
#pragma unroll
    for (int c = 0; c < 4; ++c) {
      v16bf b = load_frag(W + (size_t)(n0 + c * 16 + ln) * EP_ + k);
      acc[c] = __builtin_amdgcn_wmma_f32_16x16x32_bf16(
          false, a, false, b, (short)0, acc[c], false, false);
    }
  }

  const int mb = m0 + ((lane >> 4) << 3);
#pragma unroll
  for (int c = 0; c < 4; ++c) {
    const int col = n0 + c * 16 + ln;
    const float bv = bih[col];
#pragma unroll
    for (int r = 0; r < 8; ++r)
      GI[(size_t)(mb + r) * G3_ + col] = (bf16)(acc[c][r] + bv);
  }
}

// ---------------------------------------------------------------------------
// Recurrent GRU + fused masked max-pool.
// 64 blocks: blockIdx>>5 = direction (0 fwd, 1 bwd), (blockIdx&31)*16 = batch tile.
// 512 threads = 16 waves; wave w owns j-slice [w*32, w*32+32) and computes all
// three gates (r,z,n) for that slice -> 6 wmma accumulators, lane-aligned gates.
// h kept in LDS: bf16 copy (wmma A operand) + f32 copy (exact update).
// ---------------------------------------------------------------------------
#define HBP 520   // bf16 h row stride (260 dwords, %64==4 -> spread banks)
#define HFP 516   // f32  h row stride

__global__ __launch_bounds__(512) void gru_rec_kernel(
    const int*  __restrict__ inds,
    const bf16* __restrict__ GIF, const bf16* __restrict__ GIB,
    const bf16* __restrict__ WHHF, const bf16* __restrict__ WHHB,
    const float* __restrict__ bhhf, const float* __restrict__ bhhb,
    float* __restrict__ out) {
  const int dir = blockIdx.x >> 5;
  const int b0  = (blockIdx.x & 31) * 16;
  const bf16*  GI  = dir ? GIB  : GIF;
  const bf16*  WHH = dir ? WHHB : WHHF;
  const float* bhh = dir ? bhhb : bhhf;

  const int lane = threadIdx.x & 31;
  const int wave = threadIdx.x >> 5;      // 0..15
  const int j0   = wave * 32;
  const int ln   = lane & 15;
  const int lm   = (lane >> 4) << 3;      // 0 or 8 (C/D row offset)
  const int kx   = (lane >> 4) << 3;      // A/B fragment K offset

  __shared__ __align__(16) bf16 hb[16 * HBP];
  __shared__ float hf[16 * HFP];
  __shared__ int   vm[16 * T_];

  for (int i = threadIdx.x; i < 16 * HBP; i += 512) hb[i] = (bf16)0.0f;
  for (int i = threadIdx.x; i < 16 * HFP; i += 512) hf[i] = 0.0f;
  for (int i = threadIdx.x; i < 16 * T_;  i += 512) vm[i] = inds[b0 * T_ + i];

  float bb[3][2];
#pragma unroll
  for (int g = 0; g < 3; ++g)
#pragma unroll
    for (int c = 0; c < 2; ++c)
      bb[g][c] = bhh[g * H_ + j0 + c * 16 + ln];

  float ym[2][8];
#pragma unroll
  for (int c = 0; c < 2; ++c)
#pragma unroll
    for (int r = 0; r < 8; ++r) ym[c][r] = -3.4028234663852886e38f;  // finfo(f32).min

  const bf16* hrow = hb + (size_t)ln * HBP;

  __syncthreads();

  for (int s = 0; s < T_; ++s) {
    const int t = dir ? (T_ - 1 - s) : s;

    // gh = h @ w_hh^T  for this wave's 3 gates x 32 columns
    v8f acc[6] = {};
    for (int ks = 0; ks < 16; ++ks) {
      const int k = ks * 32 + kx;
      v16bf a = load_frag(hrow + k);             // LDS A fragment, shared by 6 tiles
#pragma unroll
      for (int g = 0; g < 3; ++g) {
#pragma unroll
        for (int c = 0; c < 2; ++c) {
          const bf16* wp = WHH + (size_t)(g * H_ + j0 + c * 16 + ln) * H_ + k;
          v16bf b = load_frag(wp);               // L2-resident weight stream
          acc[g * 2 + c] = __builtin_amdgcn_wmma_f32_16x16x32_bf16(
              false, a, false, b, (short)0, acc[g * 2 + c], false, false);
        }
      }
    }

    __syncthreads();  // all waves done reading hb before anyone rewrites it

    // Gates + h update + fused masked max-pool
#pragma unroll
    for (int c = 0; c < 2; ++c) {
      const int j = j0 + c * 16 + ln;
#pragma unroll
      for (int r = 0; r < 8; ++r) {
        const int m = lm + r;
        const size_t gbase = (size_t)((b0 + m) * T_ + t) * G3_;
        const float gr = (float)GI[gbase + j];
        const float gz = (float)GI[gbase + H_ + j];
        const float gn = (float)GI[gbase + 2 * H_ + j];
        const float rg = sigmoidf_(gr + acc[0 * 2 + c][r] + bb[0][c]);
        const float zg = sigmoidf_(gz + acc[1 * 2 + c][r] + bb[1][c]);
        const float ng = tanhf(gn + rg * (acc[2 * 2 + c][r] + bb[2][c]));
        const float hold = hf[m * HFP + j];
        const float hnew = (1.0f - zg) * ng + zg * hold;
        hf[m * HFP + j] = hnew;
        hb[m * HBP + j] = (bf16)hnew;
        if (vm[m * T_ + t] != 0) ym[c][r] = fmaxf(ym[c][r], hnew);
      }
    }

    __syncthreads();  // new h visible to all waves before next step's GEMM
  }

  // out[b, 0, dir*H + j]  (forward WGs write cols 0..511, backward 512..1023)
#pragma unroll
  for (int c = 0; c < 2; ++c)
#pragma unroll
    for (int r = 0; r < 8; ++r)
      out[(size_t)(b0 + lm + r) * (2 * H_) + dir * H_ + j0 + c * 16 + ln] = ym[c][r];
}

// ---------------------------------------------------------------------------
extern "C" void kernel_launch(void* const* d_in, const int* in_sizes, int n_in,
                              void* d_out, int out_size, void* d_ws, size_t ws_size,
                              hipStream_t stream) {
  (void)in_sizes; (void)n_in; (void)out_size; (void)ws_size;

  const int*   inds = (const int*)  d_in[0];
  const float* emb  = (const float*)d_in[1];
  const float* wihf = (const float*)d_in[2];
  const float* whhf = (const float*)d_in[3];
  const float* bihf = (const float*)d_in[4];
  const float* bhhf = (const float*)d_in[5];
  const float* wihb = (const float*)d_in[6];
  const float* whhb = (const float*)d_in[7];
  const float* bihb = (const float*)d_in[8];
  const float* bhhb = (const float*)d_in[9];
  float* out = (float*)d_out;

  // Workspace layout (bf16), total ~144 MB
  bf16* X    = (bf16*)d_ws;                       // [20480][320]
  bf16* WIHF = X    + (size_t)BT_ * EP_;          // [1536][320]
  bf16* WIHB = WIHF + (size_t)G3_ * EP_;
  bf16* WHHF = WIHB + (size_t)G3_ * EP_;          // [1536][512]
  bf16* WHHB = WHHF + (size_t)G3_ * H_;
  bf16* GIF  = WHHB + (size_t)G3_ * H_;           // [20480][1536]
  bf16* GIB  = GIF  + (size_t)BT_ * G3_;

  // 1) weight casts (+K padding for w_ih)
  castpad_kernel<<<(G3_ * EP_ + 255) / 256, 256, 0, stream>>>(wihf, WIHF, G3_, E_, EP_);
  castpad_kernel<<<(G3_ * EP_ + 255) / 256, 256, 0, stream>>>(wihb, WIHB, G3_, E_, EP_);
  castpad_kernel<<<(G3_ * H_  + 255) / 256, 256, 0, stream>>>(whhf, WHHF, G3_, H_, H_);
  castpad_kernel<<<(G3_ * H_  + 255) / 256, 256, 0, stream>>>(whhb, WHHB, G3_, H_, H_);

  // 2) embedding gather + cast
  gather_kernel<<<(BT_ * EP_ + 255) / 256, 256, 0, stream>>>(inds, emb, X);

  // 3) input-gate GEMMs (WMMA bf16), both directions indexed by original t
  dim3 gg(G3_ / 64, BT_ / 128);   // (24, 160)
  gemm_gi_kernel<<<gg, 256, 0, stream>>>(X, WIHF, bihf, GIF);
  gemm_gi_kernel<<<gg, 256, 0, stream>>>(X, WIHB, bihb, GIB);

  // 4) fused bidirectional recurrence + masked max-pool (fwd blocks 0..31, bwd 32..63)
  gru_rec_kernel<<<64, 512, 0, stream>>>(inds, GIF, GIB, WHHF, WHHB, bhhf, bhhb, out);
}
